// NeuralRoughSimulator_15762529976364
// MI455X (gfx1250) — matrix operand order repaired and checked by reading, not compiled, not run on hardware
//
#include <hip/hip_runtime.h>
#include <hip/hip_bf16.h>

// ---------------------------------------------------------------------------
// NeuralRoughSimulator on MI455X (gfx1250, wave32, WMMA)
//
// Persistent kernel: 256 blocks x 128 threads (4 waves). Each wave owns 32
// paths (2 WMMA row tiles). T=1024 sequential steps in-kernel. Weights pinned
// in LDS (f16, B-fragment layout). Drift and diffusion MLPs are interleaved
// layer-by-layer: double the independent WMMA/DS work between s_wait_dscnt,
// shared layer-0 A fragments, and WMMA hazard slots filled with real work.
// All LDS pointers are explicit address_space(3) => pure ds_* codegen.
// ---------------------------------------------------------------------------

#define AS3 __attribute__((address_space(3)))

typedef _Float16 v16h __attribute__((ext_vector_type(16)));
typedef _Float16 v8h  __attribute__((ext_vector_type(8)));
typedef _Float16 h2   __attribute__((ext_vector_type(2)));
typedef float    v8f  __attribute__((ext_vector_type(8)));

typedef AS3 _Float16 lh;       // LDS f16 element
typedef AS3 float    lf;       // LDS f32 element
typedef AS3 v8h      lv8h;     // LDS 16B vector
typedef AS3 h2       lh2;      // LDS 4B f16 pair
typedef AS3 unsigned char lbyte;

constexpr int BP = 32768;     // paths
constexpr int TS = 1024;      // timesteps
constexpr int WAVES   = 4;
constexpr int BLOCK_T = WAVES * 32;   // 128 threads
constexpr int PPB     = WAVES * 32;   // 128 paths / block

constexpr int XIN_S = 40;     // f16 stride of input buffer rows (80B, 16B mult)
constexpr int H_S   = 72;     // f16 stride of hidden buffer rows (144B, 16B mult)

constexpr float LOG_V_MIN    = -7.0f;
constexpr float LOG_V_MAX    =  2.0f;
constexpr float LOG_V_CENTER = -2.5f;
constexpr float INV_LOG_V_SCALE = 1.0f / 2.25f;
constexpr float DRIFT_SCALE  = 0.5f;
constexpr float DIFF_MIN     = 0.1f;
constexpr float DIFF_SPAN    = 1.5f;   // DIFF_MAX - DIFF_MIN

// ---- dynamic LDS layout (bytes) -------------------------------------------
constexpr unsigned OFF_W0D  = 0;                        // 64x32 f16
constexpr unsigned OFF_W1D  = OFF_W0D + 64*32*2;        // 64x64 f16
constexpr unsigned OFF_W2D  = OFF_W1D + 64*64*2;        // 64x64 f16
constexpr unsigned OFF_W3D  = OFF_W2D + 64*64*2;        // 16x64 f16 (head, padded)
constexpr unsigned OFF_W0F  = OFF_W3D + 16*64*2;
constexpr unsigned OFF_W1F  = OFF_W0F + 64*32*2;
constexpr unsigned OFF_W2F  = OFF_W1F + 64*64*2;
constexpr unsigned OFF_W3F  = OFF_W2F + 64*64*2;
constexpr unsigned OFF_BD   = OFF_W3F + 16*64*2;        // 192 f32 biases (drift)
constexpr unsigned OFF_BF   = OFF_BD  + 192*4;          // 192 f32 biases (diff)
constexpr unsigned OFF_XIN  = OFF_BF  + 192*4;          // WAVES*32*XIN_S f16
constexpr unsigned OFF_H0D  = OFF_XIN + (unsigned)WAVES*32*XIN_S*2;
constexpr unsigned OFF_H1D  = OFF_H0D + (unsigned)WAVES*32*H_S*2;
constexpr unsigned OFF_H0F  = OFF_H1D + (unsigned)WAVES*32*H_S*2;
constexpr unsigned OFF_H1F  = OFF_H0F + (unsigned)WAVES*32*H_S*2;
constexpr unsigned OFF_HEAD = OFF_H1F + (unsigned)WAVES*32*H_S*2; // WAVES*64 f32
constexpr unsigned SMEM_BYTES = OFF_HEAD + (unsigned)WAVES*64*4;  // ~131.6 KB

// ---- fast transcendental helpers (TRANS ops co-execute with XDL WMMA) -----
__device__ inline float fexp(float x) {
  return __builtin_amdgcn_exp2f(x * 1.44269504088896341f);
}
__device__ inline float flog(float x) {
  return __builtin_amdgcn_logf(x) * 0.69314718055994531f;
}
__device__ inline float ftanh(float x) {
#if __has_builtin(__builtin_amdgcn_tanhf)
  return __builtin_amdgcn_tanhf(x);          // native v_tanh_f32 (CDNA5 TRANS)
#else
  float e = __builtin_amdgcn_exp2f(x * 2.88539008177792681f); // e^{2x}
  return 1.0f - 2.0f * __builtin_amdgcn_rcpf(e + 1.0f);
#endif
}
__device__ inline float fsigmoid(float x) {
  return __builtin_amdgcn_rcpf(1.0f + __builtin_amdgcn_exp2f(-1.44269504088896341f * x));
}

// Wave-local LDS producer->consumer ordering (CDNA5 split DS counter).
__device__ inline void lds_fence() {
  asm volatile("s_wait_dscnt 0" ::: "memory");
}

__device__ inline v8f wmma32(v16h a, v16h b, v8f c) {
  return __builtin_amdgcn_wmma_f32_16x16x32_f16(false, a, false, b,
                                                (short)0, c, false, false);
}

// A fragment (16x32 f16, MxK): lane l<16 holds row m=l, K 0..7 & 16..23;
// lane l>=16 holds row m=l-16, K 8..15 & 24..31. Two ds_load_b128 per lane.
__device__ inline v16h loadA(const lh* base, int stride, int lane,
                             int rt, int kofs) {
  int m  = lane & 15;
  int kh = (lane >> 4) << 3;                       // 0 or 8
  const lh* row = base + (rt * 16 + m) * stride + kofs + kh;
  v8h lo = *(const lv8h*)(row);
  v8h hi = *(const lv8h*)(row + 16);
  v16h r;
#pragma unroll
  for (int i = 0; i < 8; ++i) { r[i] = lo[i]; r[i + 8] = hi[i]; }
  return r;
}

// B fragment (32x16 f16, KxN): lane l<16 holds column n=l, K 0..7 & 16..23;
// lane l>=16 holds K 8..15 & 24..31. Weights stored row-major [n][k] (== reference w[out][in]).
__device__ inline v16h loadB(const lh* wbase, int kpad, int lane,
                             int nt, int kt) {
  int n  = lane & 15;
  int kb = (lane >> 4) << 3;                       // 0 or 8
  const lh* p = wbase + (nt * 16 + n) * kpad + kt * 32 + kb;
  v8h lo = *(const lv8h*)(p);
  v8h hi = *(const lv8h*)(p + 16);
  v16h r;
#pragma unroll
  for (int i = 0; i < 8; ++i) { r[i] = lo[i]; r[i + 8] = hi[i]; }
  return r;
}

// C tile (16x16 f32): lane holds column n=lane%16, rows r+8*(lane/16).
// Fused bias + tanh + f16 convert + ds_store_b16 writeback.
__device__ inline void storeC_tanh(v8f c, lh* dst, int lane,
                                   int rt, int nt, float bn) {
  int n  = lane & 15;
  int rb = rt * 16 + ((lane >> 4) << 3);
#pragma unroll
  for (int r = 0; r < 8; ++r) {
    dst[(rb + r) * H_S + nt * 16 + n] = (_Float16)ftanh(c[r] + bn);
  }
}

// Dual hidden layer (drift + diff interleaved), K=64, N=64. 32 WMMAs, 1 fence.
__device__ inline void layer_k64_dual(const lh* Wd, const lf* Bld,
                                      const lh* srcd, lh* dstd,
                                      const lh* Wf, const lf* Blf,
                                      const lh* srcf, lh* dstf, int lane) {
#pragma unroll
  for (int rt = 0; rt < 2; ++rt) {
    v16h ad0 = loadA(srcd, H_S, lane, rt, 0);
    v16h ad1 = loadA(srcd, H_S, lane, rt, 32);
    v16h af0 = loadA(srcf, H_S, lane, rt, 0);
    v16h af1 = loadA(srcf, H_S, lane, rt, 32);
#pragma unroll
    for (int nt = 0; nt < 4; ++nt) {
      v16h bd0 = loadB(Wd, 64, lane, nt, 0);
      v16h bd1 = loadB(Wd, 64, lane, nt, 1);
      v8f cd = {0.f, 0.f, 0.f, 0.f, 0.f, 0.f, 0.f, 0.f};
      cd = wmma32(ad0, bd0, cd);
      cd = wmma32(ad1, bd1, cd);
      v16h bf0 = loadB(Wf, 64, lane, nt, 0);
      v16h bf1 = loadB(Wf, 64, lane, nt, 1);
      v8f cf = {0.f, 0.f, 0.f, 0.f, 0.f, 0.f, 0.f, 0.f};
      cf = wmma32(af0, bf0, cf);
      cf = wmma32(af1, bf1, cf);
      storeC_tanh(cd, dstd, lane, rt, nt, Bld[nt * 16 + (lane & 15)]);
      storeC_tanh(cf, dstf, lane, rt, nt, Blf[nt * 16 + (lane & 15)]);
    }
  }
  lds_fence();
}

// Stage one MLP's weights: f32 global -> f16 LDS, [out][k] row-major, padded.
__device__ inline void stage_mlp(lh* W0, lh* W1, lh* W2, lh* W3, lf* Bv,
                                 const float* w0, const float* b0,
                                 const float* w1, const float* b1,
                                 const float* w2, const float* b2,
                                 const float* w3, int tid) {
  for (int i = tid; i < 64 * 32; i += BLOCK_T) {
    int r = i >> 5, c = i & 31;
    W0[i] = (_Float16)((c < 15) ? w0[r * 15 + c] : 0.0f);
  }
  for (int i = tid; i < 64 * 64; i += BLOCK_T) W1[i] = (_Float16)w1[i];
  for (int i = tid; i < 64 * 64; i += BLOCK_T) W2[i] = (_Float16)w2[i];
  for (int i = tid; i < 16 * 64; i += BLOCK_T)
    W3[i] = (_Float16)((i < 64) ? w3[i] : 0.0f);
  for (int i = tid; i < 64; i += BLOCK_T) {
    Bv[i] = b0[i]; Bv[64 + i] = b1[i]; Bv[128 + i] = b2[i];
  }
}

__global__ __launch_bounds__(BLOCK_T)
void nrs_kernel(const float* __restrict__ dw0, const float* __restrict__ db0,
                const float* __restrict__ dw1, const float* __restrict__ db1,
                const float* __restrict__ dw2, const float* __restrict__ db2,
                const float* __restrict__ dw3, const float* __restrict__ db3,
                const float* __restrict__ fw0, const float* __restrict__ fb0,
                const float* __restrict__ fw1, const float* __restrict__ fb1,
                const float* __restrict__ fw2, const float* __restrict__ fb2,
                const float* __restrict__ fw3, const float* __restrict__ fb3,
                const float* __restrict__ kappa_p, const float* __restrict__ theta_p,
                const float* __restrict__ init_var, const float* __restrict__ dWm,
                const float* __restrict__ dt_p, float* __restrict__ out) {
  extern __shared__ __align__(16) unsigned char smem_raw[];
  // One explicit generic->LDS addrspace cast; everything below is pure AS(3).
  lbyte* smem = (lbyte*)smem_raw;

  const int tid  = threadIdx.x;
  const int lane = tid & 31;
  const int wave = tid >> 5;
  const int p    = blockIdx.x * PPB + tid;      // grid exactly covers B

  lh* W0d = (lh*)(smem + OFF_W0D);
  lh* W1d = (lh*)(smem + OFF_W1D);
  lh* W2d = (lh*)(smem + OFF_W2D);
  lh* W3d = (lh*)(smem + OFF_W3D);
  lh* W0f = (lh*)(smem + OFF_W0F);
  lh* W1f = (lh*)(smem + OFF_W1F);
  lh* W2f = (lh*)(smem + OFF_W2F);
  lh* W3f = (lh*)(smem + OFF_W3F);
  lf* Bd  = (lf*)(smem + OFF_BD);
  lf* Bf  = (lf*)(smem + OFF_BF);
  lh* xin = (lh*)(smem + OFF_XIN) + wave * 32 * XIN_S;
  lh* h0d = (lh*)(smem + OFF_H0D) + wave * 32 * H_S;
  lh* h1d = (lh*)(smem + OFF_H1D) + wave * 32 * H_S;
  lh* h0f = (lh*)(smem + OFF_H0F) + wave * 32 * H_S;
  lh* h1f = (lh*)(smem + OFF_H1F) + wave * 32 * H_S;
  lf* head = (lf*)(smem + OFF_HEAD) + wave * 64;   // [0..31]=drift, [32..63]=diff

  stage_mlp(W0d, W1d, W2d, W3d, Bd, dw0, db0, dw1, db1, dw2, db2, dw3, tid);
  stage_mlp(W0f, W1f, W2f, W3f, Bf, fw0, fb0, fw1, fb1, fw2, fb2, fw3, tid);
  __syncthreads();

  const float kap = kappa_p[0];
  const float th  = theta_p[0];
  const float dt  = dt_p[0];
  const float b3d = db3[0];
  const float b3f = fb3[0];

  // state (registers, one path per lane)
  float iv = init_var[p];
  iv = fminf(fmaxf(iv, fexp(LOG_V_MIN)), fexp(LOG_V_MAX));
  float log_v = flog(iv);
  float s1[2] = {0.f, 0.f};
  float s2[4] = {0.f, 0.f, 0.f, 0.f};
  float s3[8] = {0.f, 0.f, 0.f, 0.f, 0.f, 0.f, 0.f, 0.f};

  // zero the K-pad of this lane's input row once (cols 15..31 stay zero)
  lh* xr = xin + lane * XIN_S;
#pragma unroll
  for (int c = 15; c < 32; ++c) xr[c] = (_Float16)0.0f;

  const float* dWr  = dWm + (size_t)p * TS;
  float*       outr = out + (size_t)p * TS;

#pragma unroll 1
  for (int t = 0; t < TS; ++t) {
    // ---- stage net input for this wave's 32 paths (7x ds_store_b32 + b16) --
    float norm_lv = (log_v - LOG_V_CENTER) * INV_LOG_V_SCALE;
    lh2* xr2 = (lh2*)xr;
    xr2[0] = h2{(_Float16)s1[0], (_Float16)s1[1]};
    xr2[1] = h2{(_Float16)s2[0], (_Float16)s2[1]};
    xr2[2] = h2{(_Float16)s2[2], (_Float16)s2[3]};
    xr2[3] = h2{(_Float16)s3[0], (_Float16)s3[1]};
    xr2[4] = h2{(_Float16)s3[2], (_Float16)s3[3]};
    xr2[5] = h2{(_Float16)s3[4], (_Float16)s3[5]};
    xr2[6] = h2{(_Float16)s3[6], (_Float16)s3[7]};
    xr[14] = (_Float16)norm_lv;
    lds_fence();

    __builtin_prefetch(dWr + ((t + 32) & (TS - 1)), 0, 0);  // global_prefetch_b8

    // ---- layer 0 (both MLPs, shared A fragments): xin -> h0d / h0f ---------
#pragma unroll
    for (int rt = 0; rt < 2; ++rt) {
      v16h a = loadA(xin, XIN_S, lane, rt, 0);
#pragma unroll
      for (int nt = 0; nt < 4; ++nt) {
        v16h bd = loadB(W0d, 32, lane, nt, 0);
        v8f cd = {0.f, 0.f, 0.f, 0.f, 0.f, 0.f, 0.f, 0.f};
        cd = wmma32(a, bd, cd);
        v16h bf = loadB(W0f, 32, lane, nt, 0);
        v8f cf = {0.f, 0.f, 0.f, 0.f, 0.f, 0.f, 0.f, 0.f};
        cf = wmma32(a, bf, cf);
        storeC_tanh(cd, h0d, lane, rt, nt, Bd[nt * 16 + (lane & 15)]);
        storeC_tanh(cf, h0f, lane, rt, nt, Bf[nt * 16 + (lane & 15)]);
      }
    }
    lds_fence();

    // ---- layers 1 and 2 (interleaved) --------------------------------------
    layer_k64_dual(W1d, Bd + 64,  h0d, h1d, W1f, Bf + 64,  h0f, h1f, lane);
    layer_k64_dual(W2d, Bd + 128, h1d, h0d, W2f, Bf + 128, h1f, h0f, lane);

    // ---- heads (both MLPs): K=64, N=16 (col 0 real) ------------------------
#pragma unroll
    for (int rt = 0; rt < 2; ++rt) {
      v16h ad0 = loadA(h0d, H_S, lane, rt, 0);
      v16h ad1 = loadA(h0d, H_S, lane, rt, 32);
      v16h bd0 = loadB(W3d, 64, lane, 0, 0);
      v16h bd1 = loadB(W3d, 64, lane, 0, 1);
      v8f cd = {0.f, 0.f, 0.f, 0.f, 0.f, 0.f, 0.f, 0.f};
      cd = wmma32(ad0, bd0, cd);
      cd = wmma32(ad1, bd1, cd);
      v16h af0 = loadA(h0f, H_S, lane, rt, 0);
      v16h af1 = loadA(h0f, H_S, lane, rt, 32);
      v16h bf0 = loadB(W3f, 64, lane, 0, 0);
      v16h bf1 = loadB(W3f, 64, lane, 0, 1);
      v8f cf = {0.f, 0.f, 0.f, 0.f, 0.f, 0.f, 0.f, 0.f};
      cf = wmma32(af0, bf0, cf);
      cf = wmma32(af1, bf1, cf);
      if ((lane & 15) == 0) {                 // column-0 lanes own the result
        int rb = rt * 16 + ((lane >> 4) << 3);
#pragma unroll
        for (int r = 0; r < 8; ++r) { head[rb + r] = cd[r]; head[32 + rb + r] = cf[r]; }
      }
    }
    lds_fence();
    float hd = head[lane] + b3d;
    float hf = head[32 + lane] + b3f;

    // ---- SDE step + signature update (per-lane scalar) ---------------------
    float drift_nn  = DRIFT_SCALE * ftanh(hd);
    float diffusion = DIFF_SPAN * fsigmoid(hf) + DIFF_MIN;
    float dwv = dWr[t];
    float lvn = log_v + kap * (th - log_v) * dt + drift_nn * dt + diffusion * dwv;
    lvn = fminf(fmaxf(lvn, LOG_V_MIN), LOG_V_MAX);
    float d = lvn - log_v;

    float dx[2]  = {dt, d};
    float dx2[4];
#pragma unroll
    for (int i = 0; i < 2; ++i)
#pragma unroll
      for (int j = 0; j < 2; ++j) dx2[i * 2 + j] = dx[i] * dx[j];

    // s3 first (uses old s1, s2)
    float ns3[8];
#pragma unroll
    for (int i = 0; i < 4; ++i)
#pragma unroll
      for (int j = 0; j < 2; ++j) ns3[i * 2 + j] = s2[i] * dx[j];
#pragma unroll
    for (int i = 0; i < 2; ++i)
#pragma unroll
      for (int j = 0; j < 4; ++j)
        ns3[i * 4 + j] += s1[i] * (0.5f * dx2[j]) + (1.0f / 6.0f) * dx[i] * dx2[j];
#pragma unroll
    for (int k = 0; k < 8; ++k) s3[k] += ns3[k];
    // s2 (uses old s1)
#pragma unroll
    for (int i = 0; i < 2; ++i)
#pragma unroll
      for (int j = 0; j < 2; ++j)
        s2[i * 2 + j] += s1[i] * dx[j] + 0.5f * dx2[i * 2 + j];
    // s1
    s1[0] += dx[0];
    s1[1] += dx[1];

    outr[t] = fexp(lvn);      // whole output fits in 192MB L2 -> coalesces
    log_v = lvn;
  }
}

extern "C" void kernel_launch(void* const* d_in, const int* in_sizes, int n_in,
                              void* d_out, int out_size, void* d_ws, size_t ws_size,
                              hipStream_t stream) {
  (void)in_sizes; (void)n_in; (void)out_size; (void)d_ws; (void)ws_size;

  const float* dw0 = (const float*)d_in[0];
  const float* db0 = (const float*)d_in[1];
  const float* dw1 = (const float*)d_in[2];
  const float* db1 = (const float*)d_in[3];
  const float* dw2 = (const float*)d_in[4];
  const float* db2 = (const float*)d_in[5];
  const float* dw3 = (const float*)d_in[6];
  const float* db3 = (const float*)d_in[7];
  const float* fw0 = (const float*)d_in[8];
  const float* fb0 = (const float*)d_in[9];
  const float* fw1 = (const float*)d_in[10];
  const float* fb1 = (const float*)d_in[11];
  const float* fw2 = (const float*)d_in[12];
  const float* fb2 = (const float*)d_in[13];
  const float* fw3 = (const float*)d_in[14];
  const float* fb3 = (const float*)d_in[15];
  const float* kappa = (const float*)d_in[16];
  const float* theta = (const float*)d_in[17];
  const float* init_var = (const float*)d_in[18];
  const float* dWm = (const float*)d_in[19];
  const float* dtp = (const float*)d_in[20];
  float* out = (float*)d_out;

  (void)hipFuncSetAttribute(reinterpret_cast<const void*>(nrs_kernel),
                            hipFuncAttributeMaxDynamicSharedMemorySize,
                            (int)SMEM_BYTES);

  dim3 grid(BP / PPB);      // 256 blocks
  dim3 block(BLOCK_T);      // 128 threads = 4 waves
  nrs_kernel<<<grid, block, SMEM_BYTES, stream>>>(
      dw0, db0, dw1, db1, dw2, db2, dw3, db3,
      fw0, fb0, fw1, fb1, fw2, fb2, fw3, fb3,
      kappa, theta, init_var, dWm, dtp, out);
}
